// TwoWL_90924457656372
// MI455X (gfx1250) — 8-bit matrix operand order repaired and checked by reading, compiled, not run on hardware
//
#include <hip/hip_runtime.h>
#include <stdint.h>


#define NN    256
#define MM    65536
#define TSIZE 131072
#define TMASK (TSIZE - 1)
#define NONE  0x7FFFFFFF

typedef int v8i __attribute__((ext_vector_type(8)));

__global__ void k_fill(uint32_t* p, uint32_t v, int n) {
  for (int i = blockIdx.x * blockDim.x + threadIdx.x; i < n;
       i += gridDim.x * blockDim.x)
    p[i] = v;
}

__global__ void k_edges(const int* ei, int E, uint8_t* A) {
  int e = blockIdx.x * blockDim.x + threadIdx.x;
  if (e < E) A[ei[e] * NN + ei[E + e]] = 1;
}

__device__ __forceinline__ int l0_pos(int i, int j) {
  return (i == j) ? (NN * (NN - 1) + i) : (i * (NN - 1) + (j < i ? j : j - 1));
}

__global__ void k_l0_first(const int* x, const uint8_t* A, int* firstpos) {
  int p = blockIdx.x * blockDim.x + threadIdx.x;
  if (p >= MM) return;
  int i = p >> 8, j = p & 255;
  int key = (x[i] << 5) | (x[j] << 1) | (int)A[p];
  atomicMin(&firstpos[key], l0_pos(i, j));
}

__global__ void k_l0_mark(const int* firstpos, int* mark) {
  int t = blockIdx.x * blockDim.x + threadIdx.x;
  if (t < 512) {
    int fp = firstpos[t];
    if (fp != NONE) mark[fp] = 1;
  }
}

__global__ void k_l0_label(const int* x, const uint8_t* A, const int* firstpos,
                           const int* scan, int* labels, float* out0) {
  int p = blockIdx.x * blockDim.x + threadIdx.x;
  if (p >= MM) return;
  int i = p >> 8, j = p & 255;
  int key = (x[i] << 5) | (x[j] << 1) | (int)A[p];
  int lab = scan[firstpos[key]];
  labels[p] = lab;
  atomicAdd(&out0[lab], 1.0f);
}

__global__ void k_sort(const int* labels, uint8_t* hi, uint8_t* lo,
                       int colmode) {
  __shared__ unsigned s[NN];
  int r = blockIdx.x, t = threadIdx.x;
  s[t] = colmode ? (unsigned)labels[t * NN + r] : (unsigned)labels[r * NN + t];
  __syncthreads();
  for (int k = 2; k <= NN; k <<= 1) {
    for (int j = k >> 1; j > 0; j >>= 1) {
      int ixj = t ^ j;
      if (ixj > t) {
        unsigned a = s[t], b = s[ixj];
        bool up = ((t & k) == 0);
        if ((a > b) == up) { s[t] = b; s[ixj] = a; }
      }
      __syncthreads();
    }
  }
  unsigned v = s[t];
  hi[r * NN + t] = (uint8_t)(v >> 8);
  lo[r * NN + t] = (uint8_t)(v & 255);
}

__global__ void k_norms(const uint8_t* rhi, const uint8_t* rlo,
                        const uint8_t* chi, const uint8_t* clo, int* nr) {
  int id = blockIdx.x * blockDim.x + threadIdx.x;
  if (id >= 4 * NN) return;
  const uint8_t* P = (id < NN) ? rhi : (id < 2 * NN) ? rlo
                     : (id < 3 * NN) ? chi : clo;
  int r = id & 255, s = 0;
  for (int k = 0; k < NN; ++k) { int v = P[r * NN + k]; s += v * v; }
  nr[id] = s;
}

__global__ __launch_bounds__(32) void k_eq(const uint8_t* Phi,
                                           const uint8_t* Plo, const int* nhi,
                                           const int* nlo, int* rep) {
  int lane = threadIdx.x;
  int ti = blockIdx.y, tj = blockIdx.x;
  int h = lane >> 4, lm = lane & 15;
  int rowA = ti * 16 + lm;
  int colB = tj * 16 + lm;
  v8i acch = {0, 0, 0, 0, 0, 0, 0, 0};
  v8i accl = {0, 0, 0, 0, 0, 0, 0, 0};
  for (int kk = 0; kk < 4; ++kk) {
    int kb = kk * 64;
    v8i ah, al, bh, bl;
#pragma unroll
    for (int v = 0; v < 8; ++v) {
      int ka = ((v & 2) << 3) + ((v >> 2) << 5) + h * 8 + (v & 1) * 4;
      int kbo = ((v & 2) << 3) + ((v >> 2) << 5) + h * 8 + (v & 1) * 4;
      ah[v] = *(const int*)(Phi + rowA * NN + kb + ka);
      al[v] = *(const int*)(Plo + rowA * NN + kb + ka);
      bh[v] = *(const int*)(Phi + colB * NN + kb + kbo);
      bl[v] = *(const int*)(Plo + colB * NN + kb + kbo);
    }
    acch = __builtin_amdgcn_wmma_i32_16x16x64_iu8(false, ah, false, bh, acch,
                                                  false, false);
    accl = __builtin_amdgcn_wmma_i32_16x16x64_iu8(false, al, false, bl, accl,
                                                  false, false);
  }
  int j = tj * 16 + lm;
  int njh = nhi[j], njl = nlo[j];
#pragma unroll
  for (int v = 0; v < 8; ++v) {
    int i = ti * 16 + v + 8 * h;
    int dh = nhi[i] + njh - 2 * acch[v];
    int dl = nlo[i] + njl - 2 * accl[v];
    if ((dh | dl) == 0) atomicMin(&rep[i], j);
  }
}

__global__ void k_insert(const int* labels, const int* rowrep,
                         const int* colrep, unsigned long long* tkey,
                         int* tpos, int* slots) {
  int p = blockIdx.x * blockDim.x + threadIdx.x;
  if (p >= MM) return;
  int i = p >> 8, j = p & 255;
  unsigned key = ((unsigned)labels[p] << 16) | ((unsigned)rowrep[i] << 8) |
                 (unsigned)colrep[j];
  unsigned long long e = (unsigned long long)key + 1ull;
  unsigned slot = (key * 2654435761u) >> 15;
  for (;;) {
    unsigned long long prev = atomicCAS(&tkey[slot], 0ull, e);
    if (prev == 0ull || prev == e) break;
    slot = (slot + 1) & TMASK;
  }
  atomicMin(&tpos[slot], p);
  slots[p] = (int)slot;
}

__global__ void k_mark(const unsigned long long* tkey, const int* tpos,
                       int* mark) {
  int s = blockIdx.x * blockDim.x + threadIdx.x;
  if (s < TSIZE && tkey[s] != 0ull) mark[tpos[s]] = 1;
}

__global__ void k_relabel(const int* slots, const int* tpos, const int* scan,
                          int* labels, float* outl) {
  int p = blockIdx.x * blockDim.x + threadIdx.x;
  if (p >= MM) return;
  int lab = scan[tpos[slots[p]]];
  labels[p] = lab;
  atomicAdd(&outl[lab], 1.0f);
}

__global__ void k_scan1(const int* in, int* out, int* bsum) {
  __shared__ int s[256];
  int b = blockIdx.x, t = threadIdx.x;
  int v = in[b * 256 + t];
  s[t] = v;
  __syncthreads();
  for (int off = 1; off < 256; off <<= 1) {
    int x = (t >= off) ? s[t - off] : 0;
    __syncthreads();
    s[t] += x;
    __syncthreads();
  }
  out[b * 256 + t] = s[t] - v;
  if (t == 255) bsum[b] = s[255];
}

__global__ void k_scan2(const int* bsum, int* bsumEx) {
  __shared__ int s[256];
  int t = threadIdx.x;
  int v = bsum[t];
  s[t] = v;
  __syncthreads();
  for (int off = 1; off < 256; off <<= 1) {
    int x = (t >= off) ? s[t - off] : 0;
    __syncthreads();
    s[t] += x;
    __syncthreads();
  }
  bsumEx[t] = s[t] - v;
}

__global__ void k_scan3(int* out, const int* bsumEx) {
  int p = blockIdx.x * blockDim.x + threadIdx.x;
  if (p < MM) out[p] += bsumEx[p >> 8];
}

extern "C" void kernel_launch(void* const* d_in, const int* in_sizes, int n_in,
                              void* d_out, int out_size, void* d_ws,
                              size_t ws_size, hipStream_t stream) {
  (void)n_in; (void)out_size; (void)ws_size;
  const int* x  = (const int*)d_in[0];
  const int* ei = (const int*)d_in[1];
  int E = in_sizes[1] / 2;
  float* out = (float*)d_out;

  char* ws = (char*)d_ws;
  int*      labels   = (int*)(ws + 0);
  uint8_t*  A        = (uint8_t*)(ws + 262144);
  uint8_t*  rhi      = (uint8_t*)(ws + 327680);
  uint8_t*  rlo      = (uint8_t*)(ws + 393216);
  uint8_t*  chi      = (uint8_t*)(ws + 458752);
  uint8_t*  clo      = (uint8_t*)(ws + 524288);
  int*      norms    = (int*)(ws + 589824);
  int*      rowrep   = (int*)(ws + 593920);
  int*      colrep   = (int*)(ws + 594944);
  int*      first512 = (int*)(ws + 595968);
  int*      bsum     = (int*)(ws + 598016);
  int*      bsumEx   = (int*)(ws + 599040);
  int*      scanArr  = (int*)(ws + 600064);
  int*      mark     = (int*)(ws + 862208);
  int*      slots    = (int*)(ws + 1124352);
  int*      tpos     = (int*)(ws + 1386496);
  unsigned long long* tkey = (unsigned long long*)(ws + 1910784);

  k_fill<<<192, 256, 0, stream>>>((uint32_t*)out, 0u, 3 * MM);
  k_fill<<<64, 256, 0, stream>>>((uint32_t*)A, 0u, MM / 4);
  k_fill<<<2, 256, 0, stream>>>((uint32_t*)first512, (uint32_t)NONE, 512);
  k_fill<<<64, 256, 0, stream>>>((uint32_t*)mark, 0u, MM);
  k_edges<<<(E + 255) / 256, 256, 0, stream>>>(ei, E, A);
  k_l0_first<<<256, 256, 0, stream>>>(x, A, first512);
  k_l0_mark<<<2, 256, 0, stream>>>(first512, mark);
  k_scan1<<<256, 256, 0, stream>>>(mark, scanArr, bsum);
  k_scan2<<<1, 256, 0, stream>>>(bsum, bsumEx);
  k_scan3<<<256, 256, 0, stream>>>(scanArr, bsumEx);
  k_l0_label<<<256, 256, 0, stream>>>(x, A, first512, scanArr, labels, out);

  for (int l = 1; l <= 2; ++l) {
    k_sort<<<256, 256, 0, stream>>>(labels, rhi, rlo, 0);
    k_sort<<<256, 256, 0, stream>>>(labels, chi, clo, 1);
    k_norms<<<4, 256, 0, stream>>>(rhi, rlo, chi, clo, norms);
    k_fill<<<1, 256, 0, stream>>>((uint32_t*)rowrep, (uint32_t)NONE, NN);
    k_fill<<<1, 256, 0, stream>>>((uint32_t*)colrep, (uint32_t)NONE, NN);
    k_fill<<<256, 256, 0, stream>>>((uint32_t*)tkey, 0u, TSIZE * 2);
    k_fill<<<128, 256, 0, stream>>>((uint32_t*)tpos, (uint32_t)NONE, TSIZE);
    k_fill<<<64, 256, 0, stream>>>((uint32_t*)mark, 0u, MM);
    dim3 tiles(16, 16);
    k_eq<<<tiles, 32, 0, stream>>>(rhi, rlo, norms + 0, norms + 256, rowrep);
    k_eq<<<tiles, 32, 0, stream>>>(chi, clo, norms + 512, norms + 768, colrep);
    k_insert<<<256, 256, 0, stream>>>(labels, rowrep, colrep, tkey, tpos,
                                      slots);
    k_mark<<<512, 256, 0, stream>>>(tkey, tpos, mark);
    k_scan1<<<256, 256, 0, stream>>>(mark, scanArr, bsum);
    k_scan2<<<1, 256, 0, stream>>>(bsum, bsumEx);
    k_scan3<<<256, 256, 0, stream>>>(scanArr, bsumEx);
    k_relabel<<<256, 256, 0, stream>>>(slots, tpos, scanArr, labels,
                                       out + l * MM);
  }
}
